// AttentionHead_64733747085720
// MI455X (gfx1250) — matrix operand order repaired and checked by reading, not compiled, not run on hardware
//
#include <hip/hip_runtime.h>
#include <hip/hip_bf16.h>

// ---------------------------------------------------------------------------
// Single-head attention for MI455X (gfx1250, wave32, WMMA, async-LDS staging).
//   Wt  = W^T in bf16 (prep kernel)
//   q   = (Q@Wq + bq) * 1/sqrt(64)   row-major bf16
//   k   =  K@Wk + bk                 row-major bf16
//   v^T = (V@Wv + bv)^T              [D][B*S]  bf16
//   out = softmax(q k^T + mask) v    flash-fused, fp32 accumulation
// ---------------------------------------------------------------------------

typedef __attribute__((ext_vector_type(16))) __bf16 v16bf;
typedef __attribute__((ext_vector_type(8)))  __bf16 v8bf;
typedef __attribute__((ext_vector_type(8)))  float  v8f;
typedef int v4i __attribute__((vector_size(16)));   // matches builtin proto

constexpr int BATCH = 4;
constexpr int SEQ   = 4096;
constexpr int EMB   = 1024;
constexpr int HD    = 64;
constexpr int TOT   = BATCH * SEQ;

union Frag16 { v16bf v; v8bf h[2]; };

// ---- async 16B global->LDS copy (ASYNCcnt-tracked), with safe fallback ----
#if __has_builtin(__builtin_amdgcn_global_load_async_to_lds_b128)
#define HAVE_ASYNC_LDS 1
__device__ inline void cp16_g2l(const void* src, void* dst) {
    // AS1 ptr == flat address; AS3 ptr == low 32 bits of flat LDS address.
    __builtin_amdgcn_global_load_async_to_lds_b128(
        (__attribute__((address_space(1))) v4i*)(unsigned long long)src,
        (__attribute__((address_space(3))) v4i*)(unsigned int)(unsigned long long)dst,
        0, 0);
}
#define WAIT_ASYNC_8() asm volatile("s_wait_asynccnt 0x8" ::: "memory")
#define WAIT_ASYNC_0() asm volatile("s_wait_asynccnt 0x0" ::: "memory")
#else
#define HAVE_ASYNC_LDS 0
__device__ inline void cp16_g2l(const void* src, void* dst) {
    *(v8bf*)dst = *(const v8bf*)src;     // synchronous fallback
}
#define WAIT_ASYNC_8()
#define WAIT_ASYNC_0()
#endif

// ---------------------------------------------------------------------------
// Kernel 0: weight prep — W[E][D] fp32  ->  Wt[3][D][E] bf16
// grid (EMB/256, HD, 3), 256 threads.
// ---------------------------------------------------------------------------
__global__ __launch_bounds__(256) void wprep_kernel(
    const float* __restrict__ Wq, const float* __restrict__ Wk,
    const float* __restrict__ Wv, __bf16* __restrict__ Wt)
{
    const int e = blockIdx.x * 256 + threadIdx.x;
    const int d = blockIdx.y;
    const float* src = (blockIdx.z == 0) ? Wq : (blockIdx.z == 1) ? Wk : Wv;
    Wt[((size_t)blockIdx.z * HD + d) * EMB + e] = (__bf16)src[(size_t)e * HD + d];
}

// ---------------------------------------------------------------------------
// Kernel 1: projection GEMM  [B*S, E] @ W + bias -> bf16 (optionally transposed out)
// block = 128 threads (4 waves); each wave computes a 16x64 output tile.
// All fragment loads are contiguous 16B.
// ---------------------------------------------------------------------------
__global__ __launch_bounds__(128) void proj_kernel(
    const float*  __restrict__ X,     // [B*S, E] fp32
    const __bf16* __restrict__ Wt,    // [D, E]   bf16 (pre-transposed)
    const float*  __restrict__ bias,  // [D]      fp32
    float outScale, int transposedOut,
    __bf16* __restrict__ out)         // [B*S, D] or [D, B*S] bf16
{
    const int wave = threadIdx.x >> 5;
    const int lane = threadIdx.x & 31;
    const int m    = lane & 15;
    const int hi   = lane >> 4;

    const int rowBase = blockIdx.x * 64 + wave * 16;
    const float* xrow = X + (size_t)(rowBase + m) * EMB;

    v8f acc[4] = {};

    for (int k0 = 0; k0 < EMB; k0 += 32) {
        // A fragment: two contiguous groups of 8 (K = k0+8hi+[0..7], +16+[0..7])
        Frag16 a;
        const float* xa = xrow + k0 + 8 * hi;
        const float4 f0 = *(const float4*)(xa + 0);
        const float4 f1 = *(const float4*)(xa + 4);
        const float4 f2 = *(const float4*)(xa + 16);
        const float4 f3 = *(const float4*)(xa + 20);
        a.v[0] = (__bf16)f0.x;  a.v[1] = (__bf16)f0.y;
        a.v[2] = (__bf16)f0.z;  a.v[3] = (__bf16)f0.w;
        a.v[4] = (__bf16)f1.x;  a.v[5] = (__bf16)f1.y;
        a.v[6] = (__bf16)f1.z;  a.v[7] = (__bf16)f1.w;
        a.v[8]  = (__bf16)f2.x; a.v[9]  = (__bf16)f2.y;
        a.v[10] = (__bf16)f2.z; a.v[11] = (__bf16)f2.w;
        a.v[12] = (__bf16)f3.x; a.v[13] = (__bf16)f3.y;
        a.v[14] = (__bf16)f3.z; a.v[15] = (__bf16)f3.w;

        // B fragments from Wt rows: two groups of 8 (K = 8hi+[0..7], +16+[0..7])
#pragma unroll
        for (int t = 0; t < 4; ++t) {
            Frag16 b;
            const __bf16* wr = Wt + (size_t)(t * 16 + m) * EMB + k0 + 8 * hi;
            b.h[0] = *(const v8bf*)(wr);
            b.h[1] = *(const v8bf*)(wr + 16);
            acc[t] = __builtin_amdgcn_wmma_f32_16x16x32_bf16(
                false, a.v, false, b.v, (short)0, acc[t], false, false);
        }
    }

    if (!transposedOut) {
#pragma unroll
        for (int t = 0; t < 4; ++t) {
            const float bv = bias[t * 16 + m];
#pragma unroll
            for (int r = 0; r < 8; ++r) {
                const int row = rowBase + r + 8 * hi;
                out[(size_t)row * HD + t * 16 + m] =
                    (__bf16)((acc[t][r] + bv) * outScale);
            }
        }
    } else {
        // transposed: C-layout rows r=0..7 are contiguous -> one 16B store
#pragma unroll
        for (int t = 0; t < 4; ++t) {
            const float bv = bias[t * 16 + m];
            v8bf pk;
#pragma unroll
            for (int r = 0; r < 8; ++r)
                pk[r] = (__bf16)((acc[t][r] + bv) * outScale);
            *(v8bf*)(out + (size_t)(t * 16 + m) * TOT + rowBase + 8 * hi) = pk;
        }
    }
}

// ---------------------------------------------------------------------------
// Kernel 2: flash attention.
// grid = (SEQ/64, BATCH); block = 128 (4 waves, 16 q-rows each).
// K/V 64x64 tiles double-buffered in LDS via async global->LDS copies.
// ---------------------------------------------------------------------------
constexpr int LDSW = 72;   // padded LDS row stride (bf16 elems) — 144B rows

__global__ __launch_bounds__(128) void flash_kernel(
    const __bf16* __restrict__ qb,   // [B*S, D] bf16 (pre-scaled)
    const __bf16* __restrict__ kb,   // [B*S, D] bf16
    const __bf16* __restrict__ vbt,  // [D, B*S] bf16 (transposed)
    const int*    __restrict__ mask, // [1, S, S]
    float*        __restrict__ out)  // [B, S, D] fp32
{
    __shared__ __bf16 Kt[2][64][LDSW];   // [buf][key][d]
    __shared__ __bf16 Vt[2][64][LDSW];   // [buf][d][key]
    __shared__ __bf16 Pt[4][16][LDSW];   // per-wave P transpose bounce

    const int tid  = threadIdx.x;
    const int wave = tid >> 5;
    const int lane = tid & 31;
    const int m    = lane & 15;
    const int hi   = lane >> 4;

    const int bb    = blockIdx.y;
    const int qBase = blockIdx.x * 64 + wave * 16;

    const __bf16* kp = kb + (size_t)bb * SEQ * HD;

    // issue one 64-key tile of K and V^T into LDS (8 x 16B per thread)
    auto prefetch = [&](int buf, int j0) {
#pragma unroll
        for (int it = 0; it < 4; ++it) {
            const int c   = tid + 128 * it;     // 512 chunks of 16B per tile
            const int row = c >> 3;
            const int off = (c & 7) * 8;
            cp16_g2l(kp + (size_t)(j0 + row) * HD + off, &Kt[buf][row][off]);
            cp16_g2l(vbt + (size_t)row * TOT + (size_t)bb * SEQ + j0 + off,
                     &Vt[buf][row][off]);
        }
    };

    // ---- q A-fragments: two contiguous 16B global loads per frag ----
    Frag16 qa[2];
    {
        const __bf16* qrow =
            qb + ((size_t)bb * SEQ + qBase + m) * HD;
#pragma unroll
        for (int f = 0; f < 2; ++f) {
            qa[f].h[0] = *(const v8bf*)(qrow + 32 * f + 8 * hi);
            qa[f].h[1] = *(const v8bf*)(qrow + 32 * f + 8 * hi + 16);
        }
    }

    float mrow[8], lrow[8];
#pragma unroll
    for (int r = 0; r < 8; ++r) { mrow[r] = -1e30f; lrow[r] = 0.0f; }
    v8f o[4] = {};

    prefetch(0, 0);

    for (int jt = 0; jt < SEQ / 64; ++jt) {
        const int buf = jt & 1;
        const int j0  = jt * 64;
        if (jt + 1 < SEQ / 64) {
            prefetch(buf ^ 1, j0 + 64);
            WAIT_ASYNC_8();          // current tile's copies complete
        } else {
            WAIT_ASYNC_0();
        }
        __syncthreads();

        // ================= S = q @ k^T ====================================
        v8f sc[4];
#pragma unroll
        for (int t = 0; t < 4; ++t) {
            v8f z = {};
#pragma unroll
            for (int f = 0; f < 2; ++f) {
                Frag16 kf;
                const __bf16* kr = &Kt[buf][16 * t + m][32 * f + 8 * hi];
                kf.h[0] = *(const v8bf*)(kr);
                kf.h[1] = *(const v8bf*)(kr + 16);
                z = __builtin_amdgcn_wmma_f32_16x16x32_bf16(
                    false, qa[f].v, false, kf.v, (short)0, z, false, false);
            }
            sc[t] = z;
        }

        // ================= mask (reference semantics) =====================
#pragma unroll
        for (int t = 0; t < 4; ++t)
#pragma unroll
            for (int r = 0; r < 8; ++r) {
                const int qr  = qBase + r + 8 * hi;
                const int key = j0 + 16 * t + m;
                float s = sc[t][r];
                if (mask[(size_t)qr * SEQ + key] == 0) s = -__builtin_inff();
                sc[t][r] = s;
            }

        // ================= online softmax (row = 16-lane half) ============
#pragma unroll
        for (int r = 0; r < 8; ++r) {
            float tmax = fmaxf(fmaxf(sc[0][r], sc[1][r]),
                               fmaxf(sc[2][r], sc[3][r]));
#pragma unroll
            for (int off = 1; off < 16; off <<= 1)
                tmax = fmaxf(tmax, __shfl_xor(tmax, off, 32));
            const float mnew = fmaxf(mrow[r], tmax);
            const float corr = __expf(mrow[r] - mnew);
            float psum = 0.0f;
#pragma unroll
            for (int t = 0; t < 4; ++t) {
                const float p = __expf(sc[t][r] - mnew);
                sc[t][r] = p;
                psum += p;
            }
#pragma unroll
            for (int off = 1; off < 16; off <<= 1)
                psum += __shfl_xor(psum, off, 32);
            lrow[r] = lrow[r] * corr + psum;
            mrow[r] = mnew;
#pragma unroll
            for (int t = 0; t < 4; ++t) o[t][r] *= corr;
        }

        // ====== P: C-layout -> A-layout via per-wave LDS bounce ===========
#pragma unroll
        for (int t = 0; t < 4; ++t)
#pragma unroll
            for (int r = 0; r < 8; ++r)
                Pt[wave][r + 8 * hi][16 * t + m] = (__bf16)sc[t][r];
        asm volatile("s_wait_dscnt 0x0" ::: "memory");   // same-wave LDS RAW

        Frag16 pa[2];
#pragma unroll
        for (int f = 0; f < 2; ++f) {
            const __bf16* pr = &Pt[wave][m][32 * f + 8 * hi];
            pa[f].h[0] = *(const v8bf*)(pr);
            pa[f].h[1] = *(const v8bf*)(pr + 16);
        }

        // ================= O += P @ V =====================================
#pragma unroll
        for (int t = 0; t < 4; ++t) {
#pragma unroll
            for (int f = 0; f < 2; ++f) {
                Frag16 vf;
                const __bf16* vr = &Vt[buf][16 * t + m][32 * f + 8 * hi];
                vf.h[0] = *(const v8bf*)(vr);
                vf.h[1] = *(const v8bf*)(vr + 16);
                o[t] = __builtin_amdgcn_wmma_f32_16x16x32_bf16(
                    false, pa[f].v, false, vf.v, (short)0, o[t], false, false);
            }
        }
        __syncthreads();   // protect LDS buffers before next overwrite
    }

    // ---- epilogue: normalize by row sum, store fp32 ----
    float* orow = out + (size_t)bb * SEQ * HD;
#pragma unroll
    for (int t = 0; t < 4; ++t)
#pragma unroll
        for (int r = 0; r < 8; ++r) {
            const int row = qBase + r + 8 * hi;
            orow[(size_t)row * HD + 16 * t + m] = o[t][r] / lrow[r];
        }
}

// ---------------------------------------------------------------------------
extern "C" void kernel_launch(void* const* d_in, const int* in_sizes, int n_in,
                              void* d_out, int out_size, void* d_ws, size_t ws_size,
                              hipStream_t stream) {
    const float* Q    = (const float*)d_in[0];
    const float* K    = (const float*)d_in[1];
    const float* V    = (const float*)d_in[2];
    const int*   mask = (const int*)d_in[3];
    const float* Wq   = (const float*)d_in[4];
    const float* bq   = (const float*)d_in[5];
    const float* Wk   = (const float*)d_in[6];
    const float* bk   = (const float*)d_in[7];
    const float* Wv   = (const float*)d_in[8];
    const float* bv   = (const float*)d_in[9];

    // workspace: qb | kb | vbt (2 MB each) | Wt (384 KB)  -> ~6.4 MB
    __bf16* qb  = (__bf16*)d_ws;
    __bf16* kb  = qb + (size_t)TOT * HD;
    __bf16* vbt = kb + (size_t)TOT * HD;
    __bf16* Wt  = vbt + (size_t)TOT * HD;   // [3][HD][EMB]

    const dim3 blk(128);
    const float invSqrtD = 0.125f;          // 1/sqrt(64), folded into q

    wprep_kernel<<<dim3(EMB / 256, HD, 3), dim3(256), 0, stream>>>(Wq, Wk, Wv, Wt);

    const dim3 projGrid(TOT / 64);
    proj_kernel<<<projGrid, blk, 0, stream>>>(Q, Wt + 0 * HD * EMB, bq, invSqrtD, 0, qb);
    proj_kernel<<<projGrid, blk, 0, stream>>>(K, Wt + 1 * (size_t)HD * EMB, bk, 1.0f, 0, kb);
    proj_kernel<<<projGrid, blk, 0, stream>>>(V, Wt + 2 * (size_t)HD * EMB, bv, 1.0f, 1, vbt);

    flash_kernel<<<dim3(SEQ / 64, BATCH), blk, 0, stream>>>(qb, kb, vbt, mask, (float*)d_out);
}
